// ChemicalGraphConv_35914516529888
// MI455X (gfx1250) — matrix operand order repaired and checked by reading, not compiled, not run on hardware
//
#include <hip/hip_runtime.h>

typedef __attribute__((ext_vector_type(16))) _Float16 v16h;
typedef __attribute__((ext_vector_type(8)))  float    v8f;
typedef __attribute__((ext_vector_type(4)))  unsigned u32x4;
typedef __attribute__((ext_vector_type(4)))  int      i32x4;
typedef __attribute__((ext_vector_type(8)))  int      i32x8;

#define DD      128      // node feature dim
#define BOND    64
#define KCAT    320      // 2*DD + BOND
#define TM      128      // edges per block tile
#define XS_STRIDE  328   // halves per X row (pad 8, keeps 16B align)
#define MID_STRIDE 136   // halves per mid row (pad 8)
#define W1_KK   160      // KCAT/2 dwords per output column
#define W2_KK   64       // DD/2 dwords per output column

// LDS byte offsets (all 16B aligned)
#define OFF_XS   0
#define OFF_MID  (TM * XS_STRIDE * 2)                       // 83968
#define OFF_W1   (OFF_MID + TM * MID_STRIDE * 2)            // 118784
#define OFF_W2   (OFF_W1 + DD * W1_KK * 4)                  // 200704
#define OFF_GATE (OFF_W2 + DD * W2_KK * 4)                  // 233472
#define SMEM_BYTES (OFF_GATE + TM * 4)                      // 233984

static __device__ inline unsigned pack2h(float lo, float hi) {
  union { _Float16 h[2]; unsigned u; } p;
  p.h[0] = (_Float16)lo; p.h[1] = (_Float16)hi;
  return p.u;
}

// Pack W (row-major [K][128] f32) into B-fragment-friendly f16 layout:
// Wp[n * KK + kk] = {half(W[2kk][n]), half(W[2kk+1][n])}
__global__ void pack_weights_kernel(const float* __restrict__ W1,
                                    const float* __restrict__ W2,
                                    unsigned* __restrict__ W1p,
                                    unsigned* __restrict__ W2p) {
  int i = blockIdx.x * blockDim.x + threadIdx.x;
  if (i < DD * W1_KK) {
    int n = i / W1_KK, kk = i % W1_KK;
    W1p[i] = pack2h(W1[(2 * kk) * DD + n], W1[(2 * kk + 1) * DD + n]);
  }
  if (i < DD * W2_KK) {
    int n = i / W2_KK, kk = i % W2_KK;
    W2p[i] = pack2h(W2[(2 * kk) * DD + n], W2[(2 * kk + 1) * DD + n]);
  }
}

__global__ void init_out_kernel(const float* __restrict__ h,
                                float* __restrict__ out, int n) {
  int i = blockIdx.x * blockDim.x + threadIdx.x;
  if (i < n) out[i] = h[i];
}

// TDM: DMA a [nRows x rowDwords] dword tile (row-contiguous) global -> LDS.
// D# per CDNA5 ISA: group0 = {flags, lds_addr,
// global_addr, type=2}; group1 = {data_size=4B, tensor_dim0/1, tile_dim0/1,
// stride0}. Groups 2/3 zero (2D tile). Tracked on TENSORcnt.
static __device__ inline void tdm_load_2d(const void* gptr, unsigned ldsByteOff,
                                          unsigned rowDwords, unsigned nRows) {
  unsigned long long ga = (unsigned long long)gptr;
  u32x4 g0;
  g0[0] = 1u;                                        // count=1 (valid), user mode
  g0[1] = ldsByteOff;                                // lds_addr
  g0[2] = (unsigned)(ga & 0xFFFFFFFFu);              // global_addr[31:0]
  g0[3] = (unsigned)((ga >> 32) & 0x01FFFFFFu) | 0x80000000u;  // ga[56:32] | type=2
  i32x8 g1;
  g1[0] = (int)(2u << 16);                           // data_size = 4 bytes
  g1[1] = (int)((rowDwords & 0xFFFFu) << 16);        // tensor_dim0 lo
  g1[2] = (int)((rowDwords >> 16) & 0xFFFFu) |
          (int)((nRows & 0xFFFFu) << 16);            // tensor_dim0 hi | tensor_dim1 lo
  g1[3] = (int)((nRows >> 16) & 0xFFFFu) |
          (int)((rowDwords & 0xFFFFu) << 16);        // tensor_dim1 hi | tile_dim0
  g1[4] = (int)(nRows & 0xFFFFu);                    // tile_dim1 (tile_dim2=0)
  g1[5] = (int)rowDwords;                            // tensor_dim0_stride lo
  g1[6] = 0;                                         // stride0 hi | stride1 lo
  g1[7] = 0;                                         // stride1 hi
  i32x4 gz = {0, 0, 0, 0};
#if __has_builtin(__builtin_amdgcn_tensor_load_to_lds)
#if defined(__clang_major__) && (__clang_major__ >= 23)
  i32x8 gz8 = {0, 0, 0, 0, 0, 0, 0, 0};
  __builtin_amdgcn_tensor_load_to_lds(g0, g1, gz, gz, gz8, 0);
#else
  __builtin_amdgcn_tensor_load_to_lds(g0, g1, gz, gz, 0);
#endif
#endif
}

// A fragment (16x32 f16): lane<16 -> M=lane, K=k0+{0..15};
// lane>=16 -> M=lane-16, K=k0+{16..31}. Two b128 LDS loads.
static __device__ inline v16h load_a_lds(const _Float16* base, int stride,
                                         int mbase, int k0, int lane) {
  int kb = (lane < 16) ? 0 : 16;
  const _Float16* p = base + (mbase + (lane & 15)) * stride + k0 + kb;
  union { uint4 u[2]; v16h h; } f;
  f.u[0] = *(const uint4*)p;
  f.u[1] = *(const uint4*)(p + 8);
  return f.h;
}

// B fragment (32x16 f16): lane n = lane&15, K half = (lane<16 ? 0 : 16).
// Packed K-pair dwords -> 8 consecutive dwords per lane = two b128 loads.
static __device__ inline v16h load_b_lds(const unsigned* Wp, int strideDw,
                                         int ncol0, int kk0, int lane) {
  const unsigned* p = Wp + (ncol0 + (lane & 15)) * strideDw + kk0 +
                      ((lane < 16) ? 0 : 8);
  union { uint4 u[2]; v16h h; } f;
  f.u[0] = *(const uint4*)p;
  f.u[1] = *(const uint4*)(p + 4);
  return f.h;
}

__global__ void __launch_bounds__(256)
edge_mlp_kernel(const float* __restrict__ h,
                const long long* __restrict__ erow,
                const long long* __restrict__ ecol,
                const float* __restrict__ bond,
                const float* __restrict__ b1,
                const float* __restrict__ b2,
                const float* __restrict__ Wa,
                const float* __restrict__ ba,
                const unsigned* __restrict__ W1p,
                const unsigned* __restrict__ W2p,
                float* __restrict__ out, int E) {
  extern __shared__ char smem[];
  _Float16* Xs    = (_Float16*)(smem + OFF_XS);
  _Float16* Mids  = (_Float16*)(smem + OFF_MID);
  unsigned* W1s   = (unsigned*)(smem + OFF_W1);
  unsigned* W2s   = (unsigned*)(smem + OFF_W2);
  float*    gates = (float*)(smem + OFF_GATE);

  const int tid  = threadIdx.x;
  const int lane = tid & 31;
  const int wave = tid >> 5;

  // ---- stage packed weights into LDS once per persistent block ----
#if __has_builtin(__builtin_amdgcn_tensor_load_to_lds)
  if (wave == 0) {
    // Two TDM descriptors; DMA overlaps the first gather phase below.
    tdm_load_2d(W1p, OFF_W1, 2048u, 10u);   // 81920 B
    tdm_load_2d(W2p, OFF_W2, 2048u, 4u);    // 32768 B
  }
#else
  {
    const uint4* s1 = (const uint4*)W1p;
    uint4*       d1 = (uint4*)W1s;
    for (int i = tid; i < DD * W1_KK / 4; i += 256) d1[i] = s1[i];
    const uint4* s2 = (const uint4*)W2p;
    uint4*       d2 = (uint4*)W2s;
    for (int i = tid; i < DD * W2_KK / 4; i += 256) d2[i] = s2[i];
  }
#endif

  const int mt0   = (wave & 3) * 2;       // this wave's two M-tiles
  const int ncol0 = (wave >> 2) * 64;     // this wave's first output column
  const int ntiles = (E + TM - 1) / TM;

  for (int tile = blockIdx.x; tile < ntiles; tile += gridDim.x) {
    const int e0 = tile * TM;

    // ---- gather TM edges: [h[row] | h[col] | bond] -> f16 X tile ----
    for (int i = tid; i < TM * (KCAT / 4); i += 256) {
      int el = i / (KCAT / 4);
      int c  = i % (KCAT / 4);
      int e  = e0 + el;
      float4 v = make_float4(0.f, 0.f, 0.f, 0.f);
      if (e < E) {
        if (c < 32) {
          long long r = erow[e];
          v = *(const float4*)(h + r * DD + c * 4);
        } else if (c < 64) {
          long long cn = ecol[e];
          v = *(const float4*)(h + cn * DD + (c - 32) * 4);
        } else {
          v = *(const float4*)(bond + (long long)e * BOND + (c - 64) * 4);
        }
      }
      union { _Float16 hh[4]; uint2 u; } pk;
      pk.hh[0] = (_Float16)v.x; pk.hh[1] = (_Float16)v.y;
      pk.hh[2] = (_Float16)v.z; pk.hh[3] = (_Float16)v.w;
      *(uint2*)(&Xs[el * XS_STRIDE + c * 4]) = pk.u;
    }

    // ---- per-edge sigmoid gate ----
    if (tid < TM) {
      int e = e0 + tid;
      float g = 0.f;
      if (e < E) {
        float s = ba[0];
        const float* br = bond + (long long)e * BOND;
        #pragma unroll 8
        for (int j = 0; j < BOND; ++j) s += br[j] * Wa[j];
        g = 1.0f / (1.0f + expf(-s));
      }
      gates[tid] = g;
    }
#if __has_builtin(__builtin_amdgcn_tensor_load_to_lds)
    if (wave == 0) __builtin_amdgcn_s_wait_tensorcnt(0);  // no-op after iter 0
#endif
    __syncthreads();

    // ---- GEMM1: X[128x320] @ W1[320x128]; wave = 2 M-tiles x 4 N-tiles ----
    v8f acc[2][4];
    #pragma unroll
    for (int p = 0; p < 2; ++p)
      #pragma unroll
      for (int t = 0; t < 4; ++t) acc[p][t] = 0.f;
    for (int ks = 0; ks < KCAT / 32; ++ks) {
      v16h a0 = load_a_lds(Xs, XS_STRIDE, (mt0 + 0) * 16, ks * 32, lane);
      v16h a1 = load_a_lds(Xs, XS_STRIDE, (mt0 + 1) * 16, ks * 32, lane);
      #pragma unroll
      for (int t = 0; t < 4; ++t) {
        v16h b = load_b_lds(W1s, W1_KK, ncol0 + t * 16, ks * 16, lane);
        acc[0][t] = __builtin_amdgcn_wmma_f32_16x16x32_f16(
            false, a0, false, b, (short)0, acc[0][t], false, false);
        acc[1][t] = __builtin_amdgcn_wmma_f32_16x16x32_f16(
            false, a1, false, b, (short)0, acc[1][t], false, false);
      }
    }

    // ---- bias + ReLU -> f16 mid tile ----
    #pragma unroll
    for (int p = 0; p < 2; ++p) {
      #pragma unroll
      for (int t = 0; t < 4; ++t) {
        int col = ncol0 + t * 16 + (lane & 15);
        float b1v = b1[col];
        #pragma unroll
        for (int r = 0; r < 8; ++r) {
          int m = (mt0 + p) * 16 + r + ((lane < 16) ? 0 : 8);
          float v = acc[p][t][r] + b1v;
          Mids[m * MID_STRIDE + col] = (_Float16)fmaxf(v, 0.0f);
        }
      }
    }
    __syncthreads();

    // ---- GEMM2: Mid[128x128] @ W2[128x128] ----
    v8f acc2[2][4];
    #pragma unroll
    for (int p = 0; p < 2; ++p)
      #pragma unroll
      for (int t = 0; t < 4; ++t) acc2[p][t] = 0.f;
    for (int ks = 0; ks < DD / 32; ++ks) {
      v16h a0 = load_a_lds(Mids, MID_STRIDE, (mt0 + 0) * 16, ks * 32, lane);
      v16h a1 = load_a_lds(Mids, MID_STRIDE, (mt0 + 1) * 16, ks * 32, lane);
      #pragma unroll
      for (int t = 0; t < 4; ++t) {
        v16h b = load_b_lds(W2s, W2_KK, ncol0 + t * 16, ks * 16, lane);
        acc2[0][t] = __builtin_amdgcn_wmma_f32_16x16x32_f16(
            false, a0, false, b, (short)0, acc2[0][t], false, false);
        acc2[1][t] = __builtin_amdgcn_wmma_f32_16x16x32_f16(
            false, a1, false, b, (short)0, acc2[1][t], false, false);
      }
    }

    // ---- bias2, gate, atomic scatter-add into out[row[e]] ----
    #pragma unroll
    for (int p = 0; p < 2; ++p) {
      #pragma unroll
      for (int t = 0; t < 4; ++t) {
        int col = ncol0 + t * 16 + (lane & 15);
        float b2v = b2[col];
        #pragma unroll
        for (int r = 0; r < 8; ++r) {
          int m = (mt0 + p) * 16 + r + ((lane < 16) ? 0 : 8);
          int e = e0 + m;
          if (e < E) {
            float val = (acc2[p][t][r] + b2v) * gates[m];
            long long dst = erow[e];
            atomicAdd(out + dst * DD + col, val);
          }
        }
      }
    }
    __syncthreads();  // protect Xs/gates from next iteration's gather
  }
}

extern "C" void kernel_launch(void* const* d_in, const int* in_sizes, int n_in,
                              void* d_out, int out_size, void* d_ws, size_t ws_size,
                              hipStream_t stream) {
  const float*     h    = (const float*)d_in[0];
  const long long* eidx = (const long long*)d_in[1];  // int64 [2,E]
  const float*     bond = (const float*)d_in[2];
  const float*     W1   = (const float*)d_in[3];
  const float*     b1   = (const float*)d_in[4];
  const float*     W2   = (const float*)d_in[5];
  const float*     b2   = (const float*)d_in[6];
  const float*     Wa   = (const float*)d_in[7];
  const float*     ba   = (const float*)d_in[8];
  float*           out  = (float*)d_out;

  const int E  = in_sizes[1] / 2;
  const int ND = in_sizes[0];  // N * D

  unsigned* W1p = (unsigned*)d_ws;               // 128*160 dwords
  unsigned* W2p = W1p + DD * W1_KK;              // 128*64 dwords

  pack_weights_kernel<<<(DD * W1_KK + 255) / 256, 256, 0, stream>>>(W1, W2, W1p, W2p);
  init_out_kernel<<<(ND + 255) / 256, 256, 0, stream>>>(h, out, ND);

  const int ntiles  = (E + TM - 1) / TM;
  const int nblocks = ntiles < 1536 ? ntiles : 1536;  // persistent blocks
  edge_mlp_kernel<<<nblocks, 256, SMEM_BYTES, stream>>>(
      h, eidx, eidx + E, bond, b1, b2, Wa, ba, W1p, W2p, out, E);
}
